// lstm_att_with_x_att_h_57990648430704
// MI455X (gfx1250) — matrix operand order repaired and checked by reading, not compiled, not run on hardware
//
#include <hip/hip_runtime.h>
#include <hip/hip_bf16.h>
#include <stdint.h>

// ---------------- problem constants ----------------
#define Bdim 128
#define Ldim 196
#define Ddim 1024
#define ROWS (Bdim * Ldim)        // 25088 rows of A = att reshaped [B*L, D]
#define A_PITCH 1032              // bf16 elems per LDS row: 2064B, 16B-aligned, 4-bank lane stride
#define W_PITCH 1032

typedef __attribute__((ext_vector_type(16))) __bf16 v16bf;
typedef __attribute__((ext_vector_type(8)))  float  v8f;

union FragCast { uint4 q[2]; v16bf v; };

__device__ __forceinline__ unsigned short f2bf(float f) {
    unsigned int u = __float_as_uint(f);
    unsigned int r = u + 0x7FFFu + ((u >> 16) & 1u);   // round-to-nearest-even
    return (unsigned short)(r >> 16);
}

// branch-free tanh: keeps EXEC uniform, ~7 VALU ops (v_exp_f32 + rcp)
__device__ __forceinline__ float fast_tanh(float y) {
    float t = __expf(-2.0f * __builtin_fabsf(y));
    float r = (1.0f - t) * __builtin_amdgcn_rcpf(1.0f + t);
    return __builtin_copysignf(r, y);
}

// ---------------------------------------------------------------------------
// Kernel 0: convert W_a2a (f32, [D,D] row-major over d) -> bf16 workspace
// ---------------------------------------------------------------------------
__global__ void k_convert_w(const float* __restrict__ W, unsigned short* __restrict__ Wbf) {
    int i = (blockIdx.x * blockDim.x + threadIdx.x) * 4;
    float4 v = *reinterpret_cast<const float4*>(W + i);
    unsigned int p0 = (unsigned int)f2bf(v.x) | ((unsigned int)f2bf(v.y) << 16);
    unsigned int p1 = (unsigned int)f2bf(v.z) | ((unsigned int)f2bf(v.w) << 16);
    *reinterpret_cast<uint2*>(Wbf + i) = make_uint2(p0, p1);
}

// ---------------------------------------------------------------------------
// Kernel 1: hx[b,e] = h[b]·W_h2a[e] + x[b]·W_i2a[e] + b_a2a[e]+b_h2a[e]+b_i2a[e]
// ---------------------------------------------------------------------------
__global__ __launch_bounds__(256) void k_hx(
    const float* __restrict__ x,  const float* __restrict__ h,
    const float* __restrict__ Wh, const float* __restrict__ bh,
    const float* __restrict__ Wi, const float* __restrict__ bi,
    const float* __restrict__ ba, float* __restrict__ hx)
{
    __shared__ float sWh[16][33];
    __shared__ float sWi[16][33];
    __shared__ float sH[Bdim][33];
    __shared__ float sX[Bdim][33];

    const int t  = threadIdx.x;
    const int e0 = blockIdx.x * 16;
    const int el = t & 15;
    const int bg = t >> 4;

    float acc[8];
#pragma unroll
    for (int j = 0; j < 8; ++j) acc[j] = 0.f;

    for (int dc = 0; dc < Ddim; dc += 32) {
#pragma unroll
        for (int i = t; i < 16 * 32; i += 256) {
            int r = i >> 5, c = i & 31;
            sWh[r][c] = Wh[(e0 + r) * Ddim + dc + c];
            sWi[r][c] = Wi[(e0 + r) * Ddim + dc + c];
        }
#pragma unroll
        for (int i = t; i < Bdim * 32; i += 256) {
            int r = i >> 5, c = i & 31;
            sH[r][c] = h[r * Ddim + dc + c];
            sX[r][c] = x[r * Ddim + dc + c];
        }
        __syncthreads();

#pragma unroll 4
        for (int dd = 0; dd < 32; ++dd) {
            float wh = sWh[el][dd];
            float wi = sWi[el][dd];
#pragma unroll
            for (int j = 0; j < 8; ++j) {
                int b = bg + 16 * j;
                acc[j] = fmaf(wh, sH[b][dd], acc[j]);
                acc[j] = fmaf(wi, sX[b][dd], acc[j]);
            }
        }
        __syncthreads();
    }

    const int e = e0 + el;
    const float bias = ba[e] + bh[e] + bi[e];
#pragma unroll
    for (int j = 0; j < 8; ++j)
        hx[(bg + 16 * j) * Ddim + e] = acc[j] + bias;
}

// ---------------------------------------------------------------------------
// Kernel 2: fused WMMA GEMM + tanh + dot(W_d2d) -> scores[B*L]
// ---------------------------------------------------------------------------
#define A_BYTES  (Bdim * A_PITCH * 2)            // 264192
#define WP_BYTES (16 * W_PITCH * 2)              // 33024
#define HX_BYTES (2 * Ddim * 4)                  // 8192
#define WD_BYTES (Ddim * 4)                      // 4096
#define SMEM_SCORES (A_BYTES + WP_BYTES + HX_BYTES + WD_BYTES)   // 309504 <= 320KB

__global__ __launch_bounds__(256) void k_scores(
    const float* __restrict__ att, const unsigned short* __restrict__ Wbf,
    const float* __restrict__ hx,  const float* __restrict__ wd,
    const float* __restrict__ bd2d, float* __restrict__ scores)
{
    extern __shared__ char smem[];
    unsigned short* Alds  = reinterpret_cast<unsigned short*>(smem);
    unsigned short* Wlds  = reinterpret_cast<unsigned short*>(smem + A_BYTES);
    float*          hxlds = reinterpret_cast<float*>(smem + A_BYTES + WP_BYTES);
    float*          wdlds = reinterpret_cast<float*>(smem + A_BYTES + WP_BYTES + HX_BYTES);

    const int t        = threadIdx.x;
    const int lane     = t & 31;
    const int wave     = t >> 5;
    const int laneHi   = lane >> 4;
    const int nl       = lane & 15;
    const int row_base = blockIdx.x * Bdim;       // 196 blocks * 128 rows == 25088

    // ---- stage A tile: 128 rows x 1024 d, f32 -> bf16 ----
    for (int r = 0; r < Bdim; ++r) {
        float4 v = *reinterpret_cast<const float4*>(
            att + (size_t)(row_base + r) * Ddim + 4 * t);
        unsigned int p0 = (unsigned int)f2bf(v.x) | ((unsigned int)f2bf(v.y) << 16);
        unsigned int p1 = (unsigned int)f2bf(v.z) | ((unsigned int)f2bf(v.w) << 16);
        *reinterpret_cast<uint2*>(Alds + r * A_PITCH + 4 * t) = make_uint2(p0, p1);
    }
    // ---- stage hx rows (row tile spans <= 2 batches) + W_d2d ----
    const int b_first = row_base / Ldim;
    {
        int b_second = b_first + 1; if (b_second > Bdim - 1) b_second = Bdim - 1;
#pragma unroll
        for (int i = t; i < 2 * Ddim; i += 256) {
            int rr = i >> 10, cc = i & (Ddim - 1);
            hxlds[i] = hx[(rr ? b_second : b_first) * Ddim + cc];
        }
    }
#pragma unroll
    for (int i = t; i < Ddim; i += 256) wdlds[i] = wd[i];
    __syncthreads();

    const int aBase = (wave * 16 + nl) * A_PITCH + laneHi * 16;
    const int bBase = nl * W_PITCH + laneHi * 16;

    // per-accumulator hx-row selector (C layout: m = v + 8*laneHi)
    bool sel[8];
#pragma unroll
    for (int v = 0; v < 8; ++v) {
        int grow = row_base + wave * 16 + laneHi * 8 + v;
        sel[v] = (grow / Ldim) != b_first;
    }

    float sacc[8];
#pragma unroll
    for (int v = 0; v < 8; ++v) sacc[v] = 0.f;

    for (int e0 = 0; e0 < Ddim; e0 += 16) {
        // stage W panel: 16 e-rows x 1024 d of bf16
#pragma unroll
        for (int i = t; i < 2048; i += 256) {
            int r = i >> 7, c = i & 127;
            *reinterpret_cast<uint4*>(Wlds + r * W_PITCH + c * 8) =
                *reinterpret_cast<const uint4*>(Wbf + (size_t)(e0 + r) * Ddim + c * 8);
        }
        __syncthreads();

        // ---- K loop, software-pipelined fragment ping-pong ----
        v8f c = {0.f, 0.f, 0.f, 0.f, 0.f, 0.f, 0.f, 0.f};
        FragCast a0, b0, a1, b1;
        a0.q[0] = *reinterpret_cast<const uint4*>(Alds + aBase);
        a0.q[1] = *reinterpret_cast<const uint4*>(Alds + aBase + 8);
        b0.q[0] = *reinterpret_cast<const uint4*>(Wlds + bBase);
        b0.q[1] = *reinterpret_cast<const uint4*>(Wlds + bBase + 8);
#pragma unroll 4
        for (int kb = 0; kb < Ddim - 32; kb += 32) {
            const int kn = kb + 32;
            a1.q[0] = *reinterpret_cast<const uint4*>(Alds + aBase + kn);
            a1.q[1] = *reinterpret_cast<const uint4*>(Alds + aBase + kn + 8);
            b1.q[0] = *reinterpret_cast<const uint4*>(Wlds + bBase + kn);
            b1.q[1] = *reinterpret_cast<const uint4*>(Wlds + bBase + kn + 8);
            c = __builtin_amdgcn_wmma_f32_16x16x32_bf16(
                    false, a0.v, false, b0.v, (short)0, c, false, false);
            a0 = a1; b0 = b1;
        }
        c = __builtin_amdgcn_wmma_f32_16x16x32_bf16(
                false, a0.v, false, b0.v, (short)0, c, false, false);

        // ---- epilogue: tanh + weight, accumulate per-lane partials ----
        const float h0  = hxlds[e0 + nl];
        const float h1  = hxlds[Ddim + e0 + nl];
        const float wdv = wdlds[e0 + nl];
#pragma unroll
        for (int v = 0; v < 8; ++v) {
            float y = c[v] + (sel[v] ? h1 : h0);
            sacc[v] = fmaf(wdv, fast_tanh(y), sacc[v]);
        }
        __syncthreads();
    }

    // ---- one final cross-lane N-reduction + store ----
    const float bd = bd2d[0];
#pragma unroll
    for (int v = 0; v < 8; ++v) {
        float s = sacc[v];
#pragma unroll
        for (int off = 1; off < 16; off <<= 1)
            s += __shfl_xor(s, off, 32);
        if (nl == 0)
            scores[row_base + wave * 16 + laneHi * 8 + v] = s + bd;
    }
}

// ---------------------------------------------------------------------------
// Kernel 3: per-batch softmax over L, then att_res[b,d] = sum_l att[b,l,d]*w[l]
// ---------------------------------------------------------------------------
__global__ __launch_bounds__(256) void k_softmax_res(
    const float* __restrict__ att, const float* __restrict__ scores,
    float* __restrict__ out)
{
    __shared__ float red[256];
    __shared__ float wl[256];
    const int b = blockIdx.x;
    const int t = threadIdx.x;

    float s = (t < Ldim) ? scores[b * Ldim + t] : -3.0e38f;
    red[t] = s;
    __syncthreads();
#pragma unroll
    for (int o = 128; o > 0; o >>= 1) {
        if (t < o) red[t] = fmaxf(red[t], red[t + o]);
        __syncthreads();
    }
    const float mx = red[0];
    __syncthreads();

    float e = (t < Ldim) ? __expf(s - mx) : 0.f;
    red[t] = e;
    __syncthreads();
#pragma unroll
    for (int o = 128; o > 0; o >>= 1) {
        if (t < o) red[t] += red[t + o];
        __syncthreads();
    }
    const float inv = 1.0f / red[0];
    wl[t] = e * inv;
    __syncthreads();

    const float* ap = att + (size_t)b * Ldim * Ddim;
    for (int d = t; d < Ddim; d += 256) {
        float acc = 0.f;
#pragma unroll 4
        for (int l = 0; l < Ldim; ++l)
            acc = fmaf(ap[(size_t)l * Ddim + d], wl[l], acc);
        out[b * Ddim + d] = acc;
    }
}

// ---------------------------------------------------------------------------
extern "C" void kernel_launch(void* const* d_in, const int* in_sizes, int n_in,
                              void* d_out, int out_size, void* d_ws, size_t ws_size,
                              hipStream_t stream) {
    const float* x     = (const float*)d_in[0];
    const float* att   = (const float*)d_in[1];
    const float* h     = (const float*)d_in[2];
    const float* W_a2a = (const float*)d_in[3];
    const float* b_a2a = (const float*)d_in[4];
    const float* W_h2a = (const float*)d_in[5];
    const float* b_h2a = (const float*)d_in[6];
    const float* W_i2a = (const float*)d_in[7];
    const float* b_i2a = (const float*)d_in[8];
    const float* W_d2d = (const float*)d_in[9];
    const float* b_d2d = (const float*)d_in[10];
    float* out = (float*)d_out;

    char* ws = (char*)d_ws;
    unsigned short* Wbf = (unsigned short*)ws;                       // 2 MB
    float* hx     = (float*)(ws + (size_t)Ddim * Ddim * 2);          // 512 KB
    float* scores = (float*)(ws + (size_t)Ddim * Ddim * 2
                                + (size_t)Bdim * Ddim * 4);          // 100 KB

    k_convert_w<<<(Ddim * Ddim) / (256 * 4), 256, 0, stream>>>(W_a2a, Wbf);
    k_hx<<<Ddim / 16, 256, 0, stream>>>(x, h, W_h2a, b_h2a, W_i2a, b_i2a, b_a2a, hx);
    k_scores<<<ROWS / Bdim, 256, SMEM_SCORES, stream>>>(att, Wbf, hx, W_d2d, b_d2d, scores);
    k_softmax_res<<<Bdim, 256, 0, stream>>>(att, scores, out);
}
